// Attention_16209206575886
// MI455X (gfx1250) — matrix operand order repaired and checked by reading, not compiled, not run on hardware
//
#include <hip/hip_runtime.h>

#define H   1024
#define NH  16
#define DH  64
#define B_  4
#define S_  2048
#define NTOK (B_ * S_)          // 8192 tokens
#define SCALE 0.125f            // 1/sqrt(64)
#define EPS 1e-12f

typedef _Float16 v16h __attribute__((ext_vector_type(16)));
typedef float    v8f  __attribute__((ext_vector_type(8)));
typedef _Float16 h2   __attribute__((ext_vector_type(2)));
typedef float    f2   __attribute__((ext_vector_type(2)));

// ---------------------------------------------------------------------------
// Fragment loaders following CDNA5 WMMA VGPR layouts (ISA 7.12.2, wave32).
//
// A (16x32, f16): lane L holds row M = L&15.  VGPR v (halves 2v,2v+1):
//   v<4 : K = (L>=16 ? 8 : 0)  + 2v
//   v>=4: K = (L>=16 ? 24:16) + 2(v-4)
// B (32x16, f16): lane L holds column N = L&15.  VGPR v: same K as A
//   (v<4: K = (L>=16 ? 8 : 0) + 2v; v>=4: K = (L>=16 ? 24:16) + 2(v-4))
// C/D (16x16, f32): lane L, VGPR r -> element (M = r + 8*(L>=16), N = L&15)
// ---------------------------------------------------------------------------

__device__ __forceinline__ int a_k(int v, int hi) {
  return (v < 4) ? (hi * 8 + 2 * v) : (16 + hi * 8 + 2 * (v - 4));
}
__device__ __forceinline__ int b_k(int v, int hi) { return a_k(v, hi); }

// A fragment from fp32 row-major source (row stride ld), convert to f16
__device__ __forceinline__ v16h load_a_f32(const float* __restrict__ base,
                                           int ld, int lane) {
  const int row = lane & 15, hi = lane >> 4;
  const float* p = base + (size_t)row * ld;
  v16h a;
#pragma unroll
  for (int v = 0; v < 8; ++v) {
    f2 d = *(const f2*)(p + a_k(v, hi));
    a[2 * v]     = (_Float16)d.x;
    a[2 * v + 1] = (_Float16)d.y;
  }
  return a;
}

// A fragment from f16 row-major source
__device__ __forceinline__ v16h load_a_f16(const _Float16* __restrict__ base,
                                           int ld, int lane) {
  const int row = lane & 15, hi = lane >> 4;
  const _Float16* p = base + (size_t)row * ld;
  v16h a;
#pragma unroll
  for (int v = 0; v < 8; ++v) {
    h2 d = *(const h2*)(p + a_k(v, hi));
    a[2 * v] = d.x;  a[2 * v + 1] = d.y;
  }
  return a;
}

// B fragment: element (K=k, N=n) = src[n*ld + k]  (fp32 source -> f16)
__device__ __forceinline__ v16h load_b_f32(const float* __restrict__ base,
                                           int ld, int lane) {
  const int n = lane & 15, hi = lane >> 4;
  const float* p = base + (size_t)n * ld;
  v16h b;
#pragma unroll
  for (int v = 0; v < 8; ++v) {
    f2 d = *(const f2*)(p + b_k(v, hi));
    b[2 * v]     = (_Float16)d.x;
    b[2 * v + 1] = (_Float16)d.y;
  }
  return b;
}

// B fragment from f16 source
__device__ __forceinline__ v16h load_b_f16(const _Float16* __restrict__ base,
                                           int ld, int lane) {
  const int n = lane & 15, hi = lane >> 4;
  const _Float16* p = base + (size_t)n * ld;
  v16h b;
#pragma unroll
  for (int v = 0; v < 8; ++v) {
    h2 d = *(const h2*)(p + b_k(v, hi));
    b[2 * v] = d.x;  b[2 * v + 1] = d.y;
  }
  return b;
}

// A fragment from an LDS tile with row stride ld (halves) -> ds_load path
__device__ __forceinline__ v16h load_a_lds(const _Float16* tile, int ld,
                                           int lane) {
  const int row = lane & 15, hi = lane >> 4;
  const _Float16* p = tile + row * ld;
  v16h a;
#pragma unroll
  for (int v = 0; v < 8; ++v) {
    h2 d = *(const h2*)(p + a_k(v, hi));
    a[2 * v] = d.x;  a[2 * v + 1] = d.y;
  }
  return a;
}

__device__ __forceinline__ v8f wmma_f16(v16h a, v16h b, v8f c) {
  return __builtin_amdgcn_wmma_f32_16x16x32_f16(false, a, false, b,
                                                (short)0, c, false, false);
}

// max-reduction within each 16-lane half (rows live per-half in the C layout)
__device__ __forceinline__ float half16_max(float x) {
#pragma unroll
  for (int m = 1; m < 16; m <<= 1) x = fmaxf(x, __shfl_xor(x, m, 32));
  return x;
}
__device__ __forceinline__ float half16_sum(float x) {
#pragma unroll
  for (int m = 1; m < 16; m <<= 1) x += __shfl_xor(x, m, 32);
  return x;
}

// ---------------------------------------------------------------------------
// Kernel 1: out = X @ W^T + bias, f16 output.  Register-blocked 16(M)x64(N)
// per wave: one A fragment feeds 4 independent WMMA chains per k-step.
// MODE 0: token-major  out[token*H + n]              (Q, K)
// MODE 1: transposed   out[(b*H + n)*S + s]          (V -> Vt for P@V B-frags)
// grid (512, 4), block (32,4).
// ---------------------------------------------------------------------------
template <int MODE>
__global__ void __launch_bounds__(128, 1)
qkv_gemm(const float* __restrict__ X, const float* __restrict__ W,
         const float* __restrict__ bias, _Float16* __restrict__ out) {
  const int lane = threadIdx.x;
  const int m0 = blockIdx.x * 16;
  const int n0 = (blockIdx.y * 4 + threadIdx.y) * 64;   // 64 cols per wave

  const float* abase = X + (size_t)m0 * H;
  const float* bbase = W + (size_t)n0 * H;

  v8f acc[4] = {};
  for (int k0 = 0; k0 < H; k0 += 32) {
    __builtin_prefetch(abase + (size_t)(lane & 15) * H + k0 + 128, 0, 1);
    const v16h a = load_a_f32(abase + k0, H, lane);
#pragma unroll
    for (int t = 0; t < 4; ++t) {
      __builtin_prefetch(bbase + (size_t)(t * 16 + (lane & 15)) * H + k0 + 128, 0, 1);
      const v16h b = load_b_f32(bbase + (size_t)(t * 16) * H + k0, H, lane);
      acc[t] = wmma_f16(a, b, acc[t]);
    }
  }

  const int hi = lane >> 4;
#pragma unroll
  for (int t = 0; t < 4; ++t) {
    const int n = n0 + t * 16 + (lane & 15);
    const float bn = bias[n];
#pragma unroll
    for (int r = 0; r < 8; ++r) {
      const int m = m0 + r + 8 * hi;         // token index
      const float v = acc[t][r] + bn;
      if (MODE == 0) {
        out[(size_t)m * H + n] = (_Float16)v;
      } else {
        const int bb = m >> 11, s = m & (S_ - 1);
        out[(size_t)(bb * H + n) * S_ + s] = (_Float16)v;
      }
    }
  }
}

// ---------------------------------------------------------------------------
// Kernel 2: flash attention. One wave = 16 queries of one (b,h), streaming
// keys in chunks of 64 with online softmax.  Row sums are computed with a
// WMMA against an all-ones B fragment (no sum shuffles); only the row max
// uses ds_bpermute reductions.  grid (S/16, B*NH), block 32.
// __launch_bounds__(32,1): single wave per block -> full VGPR budget, no
// spills of the resident Q fragments / accumulators.
// ---------------------------------------------------------------------------
__global__ void __launch_bounds__(32, 1)
attn(const _Float16* __restrict__ Q,
     const _Float16* __restrict__ K,
     const _Float16* __restrict__ Vt,
     _Float16* __restrict__ ctx) {
  __shared__ _Float16 ptile[16][64];

  const int lane = threadIdx.x;
  const int hi = lane >> 4;
  const int q0 = blockIdx.x * 16;
  const int bh = blockIdx.y;
  const int b = bh >> 4, h = bh & 15;

  // Q fragments (reused for every key chunk): d = 0..31 and 32..63
  const _Float16* qbase = Q + ((size_t)b * S_ + q0) * H + h * DH;
  const v16h qa0 = load_a_f16(qbase, H, lane);
  const v16h qa1 = load_a_f16(qbase + 32, H, lane);

  v16h ones;
#pragma unroll
  for (int i = 0; i < 16; ++i) ones[i] = (_Float16)1.0f;

  float m_[8], l_[8];
  v8f acc[4] = {};
#pragma unroll
  for (int r = 0; r < 8; ++r) { m_[r] = -3.0e38f; l_[r] = 0.f; }

  for (int k0 = 0; k0 < S_; k0 += 64) {
    // ---- scores: four 16(query)x16(key) tiles; K-dim = DH = 64 (2 WMMAs each)
    const _Float16* kb = K + ((size_t)b * S_ + k0) * H + h * DH;
    v8f c[4];
#pragma unroll
    for (int t = 0; t < 4; ++t) {
      const v8f z = {};
      c[t] = wmma_f16(qa0, load_b_f16(kb + (size_t)(t * 16) * H, H, lane), z);
      c[t] = wmma_f16(qa1, load_b_f16(kb + (size_t)(t * 16) * H + 32, H, lane), c[t]);
    }

    // ---- online softmax per row; P -> LDS (transpose C-layout -> A-layout)
#pragma unroll
    for (int r = 0; r < 8; ++r) {
      const float s0 = c[0][r] * SCALE;
      const float s1 = c[1][r] * SCALE;
      const float s2 = c[2][r] * SCALE;
      const float s3 = c[3][r] * SCALE;
      const float mx = half16_max(fmaxf(fmaxf(s0, s1), fmaxf(s2, s3)));
      const float mn = fmaxf(m_[r], mx);
      const float corr = __expf(m_[r] - mn);
      m_[r] = mn;
      l_[r] *= corr;
      acc[0][r] *= corr; acc[1][r] *= corr;
      acc[2][r] *= corr; acc[3][r] *= corr;
      const int row = r + 8 * hi, col = lane & 15;
      ptile[row][col]      = (_Float16)__expf(s0 - mn);
      ptile[row][col + 16] = (_Float16)__expf(s1 - mn);
      ptile[row][col + 32] = (_Float16)__expf(s2 - mn);
      ptile[row][col + 48] = (_Float16)__expf(s3 - mn);
    }

    // ---- P fragments (16x32 each) from LDS
    const v16h pa0 = load_a_lds(&ptile[0][0], 64, lane);
    const v16h pa1 = load_a_lds(&ptile[0][0] + 32, 64, lane);

    // ---- row sums via WMMA against all-ones B (replaces sum shuffles)
    {
      v8f ls = {};
      ls = wmma_f16(pa0, ones, ls);
      ls = wmma_f16(pa1, ones, ls);
#pragma unroll
      for (int r = 0; r < 8; ++r) l_[r] += ls[r];
    }

    // ---- ctx += P(16x64) @ V(64x64): B from transposed V
    const _Float16* vb = Vt + ((size_t)(b * H) + h * DH) * S_ + k0;
#pragma unroll
    for (int t = 0; t < 4; ++t) {
      acc[t] = wmma_f16(pa0, load_b_f16(vb + (size_t)(t * 16) * S_,      S_, lane), acc[t]);
      acc[t] = wmma_f16(pa1, load_b_f16(vb + (size_t)(t * 16) * S_ + 32, S_, lane), acc[t]);
    }
  }

  // ---- normalize and emit ctx (token-major f16, [token, h*64+d])
#pragma unroll
  for (int r = 0; r < 8; ++r) {
    const float inv = 1.0f / l_[r];
    const int token = b * S_ + q0 + r + 8 * hi;
    _Float16* o = ctx + (size_t)token * H + h * DH + (lane & 15);
    o[0]  = (_Float16)(acc[0][r] * inv);
    o[16] = (_Float16)(acc[1][r] * inv);
    o[32] = (_Float16)(acc[2][r] * inv);
    o[48] = (_Float16)(acc[3][r] * inv);
  }
}

// ---------------------------------------------------------------------------
// Kernel 3: out = LayerNorm(ctx @ Wo^T + bo + X) * gamma + beta
// grid 512, block 256 (8 waves): 16 tokens per block.  Each wave computes a
// 16(M)x128(N) strip: one A fragment feeds 8 independent WMMA chains.
// ---------------------------------------------------------------------------
__global__ void __launch_bounds__(256, 1)
out_ln(const _Float16* __restrict__ Ctx,
       const float* __restrict__ Wo, const float* __restrict__ bo,
       const float* __restrict__ X,
       const float* __restrict__ gamma,
       const float* __restrict__ beta,
       float* __restrict__ out) {
  __shared__ float hbuf[16][H];          // 64 KB

  const int tid = threadIdx.x;
  const int lane = tid & 31;
  const int wave = tid >> 5;
  const int hi = lane >> 4;
  const int m0 = blockIdx.x * 16;

  // GEMM: wave's N strip = [wave*128, wave*128+128)
  const int nbase = wave * 128;
  const _Float16* abase = Ctx + (size_t)m0 * H;
  const float* bbase = Wo + (size_t)nbase * H;

  v8f acc[8] = {};
  for (int k0 = 0; k0 < H; k0 += 32) {
    __builtin_prefetch(abase + (size_t)(lane & 15) * H + k0 + 128, 0, 1);
    const v16h a = load_a_f16(abase + k0, H, lane);
#pragma unroll
    for (int t = 0; t < 8; ++t) {
      const v16h b = load_b_f32(bbase + (size_t)(t * 16) * H + k0, H, lane);
      acc[t] = wmma_f16(a, b, acc[t]);
    }
  }
#pragma unroll
  for (int t = 0; t < 8; ++t) {
    const int n = nbase + t * 16 + (lane & 15);
    const float bn = bo[n];
#pragma unroll
    for (int r = 0; r < 8; ++r) hbuf[r + 8 * hi][n] = acc[t][r] + bn;
  }
  __syncthreads();

  // Residual + LayerNorm: 16 threads per token
  const int m = tid >> 4;
  const int c = tid & 15;
  const int token = m0 + m;
  float s = 0.f, s2 = 0.f;
  for (int j = c; j < H; j += 16) {
    const float x = hbuf[m][j] + X[(size_t)token * H + j];
    hbuf[m][j] = x;
    s += x; s2 += x * x;
  }
  s = half16_sum(s);
  s2 = half16_sum(s2);
  const float mean = s * (1.0f / H);
  const float var = s2 * (1.0f / H) - mean * mean;
  const float rstd = rsqrtf(var + EPS);
  for (int j = c; j < H; j += 16) {
    const float xn = (hbuf[m][j] - mean) * rstd;
    out[(size_t)token * H + j] = gamma[j] * xn + beta[j];
  }
}

// ---------------------------------------------------------------------------
extern "C" void kernel_launch(void* const* d_in, const int* in_sizes, int n_in,
                              void* d_out, int out_size, void* d_ws, size_t ws_size,
                              hipStream_t stream) {
  (void)in_sizes; (void)n_in; (void)out_size; (void)ws_size;

  const float* X     = (const float*)d_in[0];
  const float* Wq    = (const float*)d_in[1];
  const float* bq    = (const float*)d_in[2];
  const float* Wk    = (const float*)d_in[3];
  const float* bk    = (const float*)d_in[4];
  const float* Wv    = (const float*)d_in[5];
  const float* bv    = (const float*)d_in[6];
  const float* Wo    = (const float*)d_in[7];
  const float* bo    = (const float*)d_in[8];
  const float* gamma = (const float*)d_in[9];
  const float* beta  = (const float*)d_in[10];
  float* out = (float*)d_out;

  const size_t TEN = (size_t)NTOK * H * sizeof(_Float16);   // 16 MB each
  char* ws = (char*)d_ws;
  _Float16* qf = (_Float16*)(ws);
  _Float16* kf = (_Float16*)(ws + TEN);
  _Float16* vt = (_Float16*)(ws + 2 * TEN);
  _Float16* cx = (_Float16*)(ws + 3 * TEN);

  const dim3 gGemm(NTOK / 16, H / 256);     // (512, 4)
  const dim3 bGemm(32, 4);
  qkv_gemm<0><<<gGemm, bGemm, 0, stream>>>(X, Wq, bq, qf);
  qkv_gemm<0><<<gGemm, bGemm, 0, stream>>>(X, Wk, bk, kf);
  qkv_gemm<1><<<gGemm, bGemm, 0, stream>>>(X, Wv, bv, vt);

  attn<<<dim3(S_ / 16, B_ * NH), dim3(32), 0, stream>>>(qf, kf, vt, cx);

  out_ln<<<dim3(NTOK / 16), dim3(256), 0, stream>>>(cx, Wo, bo, X, gamma, beta, out);
}
